// Mymodel_90348932038960
// MI455X (gfx1250) — compile-verified, matrix operand order checked by reading, not run on hardware
//
#include <hip/hip_runtime.h>
#include <hip/hip_bf16.h>

// ---------------------------------------------------------------------------
// pvig-style GNN forward for MI455X (gfx1250, wave32).
//   - all GEMM-shaped work through one fused WMMA kernel
//     (v_wmma_f32_16x16x32_f16, f32 accum)
//   - A tiles double-buffered in LDS, fed by the Tensor Data Mover
//     (tensor_load_to_lds pipelined against compute via s_wait_tensorcnt 1)
//   - hot 1x1 chains consume f16 B matrices pre-swizzled into the exact
//     WMMA B-fragment layout -> one contiguous 32B load per fragment,
//     4 accumulators per wave
//   - BN scale/shift + exact GELU + residual fused into the epilogue
// ---------------------------------------------------------------------------

#define KNN 9

typedef __attribute__((ext_vector_type(16))) _Float16 v16h;
typedef __attribute__((ext_vector_type(8)))  float    v8f;
typedef unsigned int u32x4 __attribute__((ext_vector_type(4)));
typedef int          i32x4 __attribute__((ext_vector_type(4)));
typedef int          i32x8 __attribute__((ext_vector_type(8)));

#if defined(__gfx1250__) && __has_builtin(__builtin_amdgcn_tensor_load_to_lds) && \
    __has_builtin(__builtin_amdgcn_s_wait_tensorcnt)
#define GEMM_TDM 1
#else
#define GEMM_TDM 0
#endif

__device__ __forceinline__ float gelu_f(float x) {
  return 0.5f * x * (1.0f + erff(x * 0.70710678118654752f));
}

// fragment-swizzled f16 layout: element (k,n) of a [K x Ng] B matrix lives at
//   ((k/32)*Ng + n)*32 + hi*16 + j,  hi=(k>>3)&1, j=(k&7)+((k>>4)<<3)
// so a lane's 16 halves for one 16x32 fragment are contiguous (32 bytes).
__device__ __forceinline__ long long sw_off(int k, long long n, long long Ng) {
  const int kr = k & 31;
  const int hib = (kr >> 3) & 1;
  const int jj = (kr & 7) + ((kr >> 4) << 3);
  return ((((long long)(k >> 5) * Ng + n) * 2 + hib) << 4) + jj;
}

#if GEMM_TDM
// Issue one TDM descriptor: 2D tile 32(k) x 16(m) of f32 -> LDS, row stride
// padded to 34 dwords; rem = remaining K (hardware zero-fills the tail).
__device__ __forceinline__ void tdm_issue_a(const float* gsrc, unsigned lds,
                                            int rem, int K) {
  const unsigned long long ga = (unsigned long long)(uintptr_t)gsrc;
  u32x4 g0;
  g0[0] = 1u;                                   // count=1, load descriptor
  g0[1] = lds;                                  // lds_addr
  g0[2] = (unsigned)(ga & 0xFFFFFFFFu);         // global_addr lo
  g0[3] = (unsigned)((ga >> 32) & 0x01FFFFFFu) | 0x80000000u;  // hi | type=2
  i32x8 g1;
  g1[0] = (int)(0x20000u | (1u << 20) | (4u << 22) | (1u << 25));
           // data_size=4B | pad_enable | interval=32dw | amount=2dw
  g1[1] = (int)(((unsigned)rem & 0xFFFFu) << 16);        // tensor_dim0 lo
  g1[2] = (int)(((unsigned)rem >> 16) | (16u << 16));    // dim0 hi | dim1=16
  g1[3] = (int)(32u << 16);                              // tile_dim0=32
  g1[4] = 16;                                            // tile_dim1=16
  g1[5] = (int)(unsigned)K;                              // tensor_dim0_stride
  g1[6] = 0; g1[7] = 0;
  const i32x4 gz = {0, 0, 0, 0};
#if __clang_major__ >= 23
  const i32x8 gz8 = {0, 0, 0, 0, 0, 0, 0, 0};
  __builtin_amdgcn_tensor_load_to_lds(g0, g1, gz, gz, gz8, 0);
#else
  __builtin_amdgcn_tensor_load_to_lds(g0, g1, gz, gz, 0);
#endif
}
#endif

// ---------------------------------------------------------------------------
// Fused GEMM: D[m,n] = sum_k A[m,k] * B[k,n]  (per z-slice)
//   MODE 0: B = implicit im2col of f32 NCHW input, 3x3 kernel
//   MODE 1: B = 1x1/linear gather from f32 NCHW input (K % 32 == 0)
//   MODE 2: B = fragment-swizzled f16 (one v16h load per fragment)
// 4 waves / 128 threads; NT 16x16 accumulators per wave (NT=4 for MODE 2).
// ---------------------------------------------------------------------------
struct GP {
  const float*    A;
  const float*    Bf;
  const _Float16* Bh;      // MODE 2 source (fragment-swizzled)
  const float*    scale;   // null -> no scale
  const float*    shift;   // used as plain bias if scale==null
  const float*    res;     // optional residual (same layout as output)
  float*          C;
  _Float16*       h16;     // optional fragment-swizzled f16 output
  int M, N, K;
  int aMod; long long aStride;
  long long bStrideZ;
  int Cin, IH, IW, strd, pad, OW, OHW;  // gather params (MODE 0/1)
  int outMode;             // 0 = NCHW via n-decomposition, 1 = plain row-major
  long long cStrideZ;      // outMode=1 z stride
  int gelu;
};

template <int MODE, int NT>
__global__ __launch_bounds__(128) void gemm_bn_act(GP p) {
  __shared__ float As[2][16 * 34];         // double-buffered, 34-dword rows
  const int lane = threadIdx.x & 31;
  const int wave = threadIdx.x >> 5;
  const int lo = lane & 15;
  const int hi = lane >> 4;
  const int m0 = blockIdx.y * 16;
  const int z  = blockIdx.z;
  const float* A = p.A + (long long)(z % p.aMod) * p.aStride;
  const _Float16* Bh = p.Bh + (long long)z * p.bStrideZ;
  const int IHW = p.IH * p.IW;

  // hoist all n-dependent decomposition out of the K loop
  int nn[NT]; long long inBase[NT]; int ohv[NT], owv[NT];
#pragma unroll
  for (int t = 0; t < NT; ++t) {
    const int n = (blockIdx.x * (4 * NT) + wave * NT + t) * 16 + lo;
    nn[t] = n;
    const int nc = (n < p.N) ? n : (p.N - 1);    // clamp: loads stay in-bounds
    if (MODE == 2) {
      inBase[t] = nc;
    } else {
      const int bimg = nc / p.OHW;
      const int hw = nc - bimg * p.OHW;
      const int oh = hw / p.OW, ow = hw - oh * p.OW;
      if (MODE == 1) {
        inBase[t] = (long long)bimg * p.Cin * IHW
                  + (long long)(oh * p.strd) * p.IW + ow * p.strd;
      } else {
        ohv[t] = oh; owv[t] = ow;
        inBase[t] = (long long)bimg * p.Cin * IHW;
      }
    }
  }

  v8f acc[NT] = {};
#if GEMM_TDM
  if (wave == 0)   // prologue: start DMA of the first A tile
    tdm_issue_a(A + (long long)m0 * p.K,
                (unsigned)(unsigned long long)(uintptr_t)&As[0][0], p.K, p.K);
#endif

  for (int k0 = 0; k0 < p.K; k0 += 32) {
    const int ib = (k0 >> 5) & 1;
#if GEMM_TDM
    if (wave == 0) {
      if (k0 + 32 < p.K) {   // pipeline: DMA next tile, wait only for current
        tdm_issue_a(A + (long long)m0 * p.K + k0 + 32,
                    (unsigned)(unsigned long long)(uintptr_t)&As[ib ^ 1][0],
                    p.K - (k0 + 32), p.K);
        __builtin_amdgcn_s_wait_tensorcnt(1);
      } else {
        __builtin_amdgcn_s_wait_tensorcnt(0);
      }
    }
#else
    for (int i = threadIdx.x; i < 512; i += 128) {
      const int kk = i & 31, mm = i >> 5;
      As[ib][mm * 34 + kk] =
          (k0 + kk < p.K) ? A[(long long)(m0 + mm) * p.K + k0 + kk] : 0.f;
    }
#endif
    __syncthreads();

    // ---- A fragment: ISA 16-bit layout, k(j) = (j&7) + 8*hi + 16*(j>>3) ----
    v16h a;
#pragma unroll
    for (int j = 0; j < 16; ++j) {
      const int kj = (j & 7) + (hi << 3) + ((j >> 3) << 4);
      a[j] = (_Float16)As[ib][lo * 34 + kj];
    }
    // ---- B fragments ----
    v16h bt[NT];
#pragma unroll
    for (int t = 0; t < NT; ++t) {
      if (MODE == 2) {
        const _Float16* bp =
            Bh + ((((long long)(k0 >> 5) * p.N + inBase[t]) * 2 + hi) << 4);
        bt[t] = *(const v16h*)bp;                    // 2x global_load_b128
        if (k0 + 32 < p.K)
          __builtin_prefetch(bp + 2ll * p.N * 16, 0, 1);
      } else {
        v16h b;
#pragma unroll
        for (int j = 0; j < 16; ++j) {
          const int kj = (j & 7) + (hi << 3) + ((j >> 3) << 4);
          const int k = k0 + kj;
          if (MODE == 1) {
            b[j] = (_Float16)p.Bf[inBase[t] + (long long)k * IHW];
          } else {
            const int ci = k / 9;
            const int r2 = k - ci * 9;
            const int kh = r2 / 3, kw = r2 - kh * 3;
            const int ih = ohv[t] * p.strd - p.pad + kh;
            const int iw = owv[t] * p.strd - p.pad + kw;
            float v = 0.f;
            if (k < p.K && (unsigned)ih < (unsigned)p.IH && (unsigned)iw < (unsigned)p.IW)
              v = p.Bf[inBase[t] + (long long)ci * IHW + ih * p.IW + iw];
            b[j] = (_Float16)v;
          }
        }
        bt[t] = b;
      }
    }
#pragma unroll
    for (int t = 0; t < NT; ++t)
      acc[t] = __builtin_amdgcn_wmma_f32_16x16x32_f16(false, a, false, bt[t],
                                                      (short)0, acc[t],
                                                      false, false);
    __syncthreads();
  }

  // ---- epilogue: C/D layout — lane lo = n, row m = r + 8*hi ----
#pragma unroll
  for (int t = 0; t < NT; ++t) {
    const int n = nn[t];
    if (n >= p.N) continue;
#pragma unroll
    for (int r = 0; r < 8; ++r) {
      const int m = m0 + r + (hi << 3);
      const int ch = (z % p.aMod) * p.M + m;
      float y = acc[t][r];
      if (p.scale)      y = y * p.scale[ch] + p.shift[ch];
      else if (p.shift) y = y + p.shift[ch];
      if (p.gelu)       y = gelu_f(y);
      long long off;
      if (p.outMode == 0) {
        const int bimg = n / p.OHW;
        const int hw = n - bimg * p.OHW;
        off = ((long long)bimg * p.M + m) * p.OHW + hw;
      } else {
        off = (long long)z * p.cStrideZ + (long long)m * p.N + n;
      }
      if (p.res) y += p.res[off];
      p.C[off] = y;
      if (p.h16) p.h16[sw_off(m, n, p.N)] = (_Float16)y;   // next GEMM's B
    }
  }
}

// ---------------------------------------------------------------------------
// Small elementwise / reduction kernels
// ---------------------------------------------------------------------------
__global__ void bn_prep(const float* g, const float* b, const float* m,
                        const float* v, const float* cb,
                        float* sc, float* sh, int C) {
  int c = blockIdx.x * 256 + threadIdx.x;
  if (c >= C) return;
  float s = g[c] * rsqrtf(v[c] + 1e-5f);
  sc[c] = s;
  sh[c] = ((cb ? cb[c] : 0.f) - m[c]) * s + b[c];
}

// x[(b,c,hw)] += pos[(c,hw)]; also emit fragment-swizzled f16 (k=c, n=b*hw+s)
__global__ void add_bcast_sw(float* x, const float* p, _Float16* xsw,
                             long long total, int chw, int hw) {
  long long t = (long long)blockIdx.x * 256 + threadIdx.x;
  if (t >= total) return;
  float y = x[t] + p[t % chw];
  x[t] = y;
  const int s = (int)(t % hw);
  long long t2 = t / hw;
  const int C = chw / hw;
  const int c = (int)(t2 % C);
  const int b = (int)(t2 / C);
  xsw[sw_off(c, (long long)b * hw + s, (long long)32 * hw)] = (_Float16)y;
}

__global__ void mean_hw(const float* x, float* o, int bc, int hw) {
  int t = blockIdx.x * 256 + threadIdx.x;
  if (t >= bc) return;
  const float* xp = x + (long long)t * hw;
  float s = 0.f;
  for (int i = 0; i < hw; ++i) s += xp[i];
  o[t] = s / (float)hw;
}

__global__ void avgpool_k(const float* x, float* y, int bc, int H, int W, int r) {
  int MW = W / r, MH = H / r, M = MH * MW;
  int t = blockIdx.x * 256 + threadIdx.x;
  if (t >= bc * M) return;
  int b = t / M, mm = t - b * M;
  int mh = mm / MW, mw = mm - mh * MW;
  const float* xp = x + (long long)b * H * W;
  float s = 0.f;
  for (int i = 0; i < r; ++i)
    for (int j = 0; j < r; ++j)
      s += xp[(mh * r + i) * W + mw * r + j];
  y[t] = s / (float)(r * r);
}

__global__ void col_norms(const float* x, float* s, float* rr, int B, int C, int N) {
  int t = blockIdx.x * 256 + threadIdx.x;
  if (t >= B * N) return;
  int b = t / N, n = t - b * N;
  float acc = 0.f;
  for (int c = 0; c < C; ++c) {
    float v = x[((long long)b * C + c) * N + n];
    acc += v * v;
  }
  s[t] = acc;
  rr[t] = rsqrtf(fmaxf(acc, 1e-24f));
}

__global__ void knn_topk(const float* feat, const float* y, const float* relpos,
                         const float* sx, const float* rx,
                         const float* sy, const float* ry,
                         int B, int C, int N, int M, int* idx) {
  int t = blockIdx.x * 256 + threadIdx.x;
  if (t >= B * N) return;
  int b = t / N, n = t - b * N;
  float bd[KNN]; int bi[KNN];
#pragma unroll
  for (int k = 0; k < KNN; ++k) { bd[k] = 3.4e38f; bi[k] = 0; }
  float rxn = rx[t];
  float sxn = sx[t] * rxn * rxn;
  for (int m = 0; m < M; ++m) {
    float dot = 0.f;
    for (int c = 0; c < C; ++c)
      dot += feat[((long long)b * C + c) * N + n] * y[((long long)b * C + c) * M + m];
    float rym = ry[b * M + m];
    float d = sxn - 2.f * dot * rxn * rym + sy[b * M + m] * rym * rym
            + relpos[(long long)n * M + m];
    if (d < bd[KNN - 1]) {               // strict < keeps earliest index on ties
      int pos = KNN - 1;
      while (pos > 0 && d < bd[pos - 1]) {
        bd[pos] = bd[pos - 1]; bi[pos] = bi[pos - 1]; --pos;
      }
      bd[pos] = d; bi[pos] = m;
    }
  }
  for (int k = 0; k < KNN; ++k) idx[(long long)t * KNN + k] = bi[k];
}

// EdgeConv messages written directly in fragment-swizzled layout, z=(b,g)
__global__ void gather_msg(const float* feat, const float* y, const int* idx,
                           _Float16* msg, int B, int C, int N, int M) {
  long long t = (long long)blockIdx.x * 256 + threadIdx.x;
  long long total = (long long)B * 2 * C * N * KNN;
  if (t >= total) return;
  int k = (int)(t % KNN); long long t2 = t / KNN;
  int n = (int)(t2 % N); t2 /= N;
  int cm = (int)(t2 % (2 * C));
  int b = (int)(t2 / (2 * C));
  float v;
  if (cm < C) {
    v = feat[((long long)b * C + cm) * N + n];
  } else {
    int c2 = cm - C;
    int j = idx[((long long)b * N + n) * KNN + k];
    v = y[((long long)b * C + c2) * M + j] - feat[((long long)b * C + c2) * N + n];
  }
  const int Cg = C / 2;                 // 2C/4
  const int g = cm / Cg;
  const int r = cm - g * Cg;
  const long long NK = (long long)N * KNN;
  msg[((long long)(b * 4 + g)) * Cg * NK + sw_off(r, (long long)n * KNN + k, NK)]
      = (_Float16)v;
}

// max over K; emits f32 (B,2C,N) + fragment-swizzled f16 (k=ch, n=b*N+n)
__global__ void maxk_k(const float* h, float* o, _Float16* osw,
                       int C2, int N, long long total) {
  long long t = (long long)blockIdx.x * 256 + threadIdx.x;
  if (t >= total) return;
  const float* hp = h + t * KNN;
  float m = hp[0];
#pragma unroll
  for (int k = 1; k < KNN; ++k) m = fmaxf(m, hp[k]);
  o[t] = m;
  const int n = (int)(t % N);
  long long t2 = t / N;
  const int ch = (int)(t2 % C2);
  const int b = (int)(t2 / C2);
  osw[sw_off(ch, (long long)b * N + n, (long long)32 * N)] = (_Float16)m;
}

__global__ void gru_add_k(const float* h, const float* gx, const float* gh,
                          float* o, int B, int L) {
  int t = blockIdx.x * 256 + threadIdx.x;
  if (t >= B * L) return;
  int b = t / L, l = t - b * L;
  float ir = gx[b * 3 * L + l], ii = gx[b * 3 * L + L + l], inn = gx[b * 3 * L + 2 * L + l];
  float hr = gh[b * 3 * L + l], hi = gh[b * 3 * L + L + l], hn  = gh[b * 3 * L + 2 * L + l];
  float r = 1.f / (1.f + expf(-(ir + hr)));
  float z = 1.f / (1.f + expf(-(ii + hi)));
  float nn = tanhf(inn + r * hn);
  float h0 = h[t];
  o[t] = (1.f - z) * nn + z * h0 + h0;   // gru_f(...) + xg
}

// ---------------------------------------------------------------------------
// Host: parse params (jax pytree order: dicts sorted by key), run the graph.
// ---------------------------------------------------------------------------
extern "C" void kernel_launch(void* const* d_in, const int* in_sizes, int n_in,
                              void* d_out, int out_size, void* d_ws, size_t ws_size,
                              hipStream_t stream) {
  (void)in_sizes; (void)n_in; (void)out_size; (void)ws_size;
  int ti = 0;
  auto nx = [&]() -> const float* { return (const float*)d_in[ti++]; };
  const float* input = nx();                 // (32,1,128,128)

  struct BNp { const float *b, *g, *m, *v; };
  struct CVp { const float *b, *w; };
  auto rdbn = [&]() { BNp r; r.b = nx(); r.g = nx(); r.m = nx(); r.v = nx(); return r; };
  auto rdcv = [&](bool bias) { CVp r; r.b = bias ? nx() : nullptr; r.w = nx(); return r; };

  struct BlkP {
    BNp f1bn; CVp f1; BNp f2bn; CVp f2;                            // ffn
    BNp gf1bn; CVp gf1; BNp gf2bn; CVp gf2; BNp gbn; CVp gconv;    // grapher
    const float* relpos;
  };
  auto rdblk = [&]() {
    BlkP x;
    x.f1bn = rdbn();  x.f1 = rdcv(true);  x.f2bn = rdbn();  x.f2 = rdcv(true);
    x.gf1bn = rdbn(); x.gf1 = rdcv(true); x.gf2bn = rdbn(); x.gf2 = rdcv(true);
    x.gbn = rdbn();   x.gconv = rdcv(true); x.relpos = nx();
    return x;
  };
  struct DSp { BNp bn; CVp cv; };
  auto rdds = [&]() { DSp r; r.bn = rdbn(); r.cv = rdcv(true); return r; };

  BlkP bk0 = rdblk(), bk1 = rdblk(); DSp dn2 = rdds();
  BlkP bk3 = rdblk(), bk4 = rdblk(); DSp dn5 = rdds();
  BlkP bk6 = rdblk(), bk7 = rdblk();
  CVp con0 = rdcv(true), con1 = rdcv(true), con2 = rdcv(true);
  struct GRp { CVp h2h, x2h; };
  GRp gr0, gr1, gr2;
  gr0.h2h = rdcv(true); gr0.x2h = rdcv(true);
  gr1.h2h = rdcv(true); gr1.x2h = rdcv(true);
  gr2.h2h = rdcv(true); gr2.x2h = rdcv(true);
  BNp pf_b1 = rdbn(), pf_b2 = rdbn(); CVp pf_c1 = rdcv(true), pf_c2 = rdcv(true);
  const float* pos_embed = nx();
  CVp pred0 = rdcv(true), pred1 = rdcv(true);
  BNp s10bn = rdbn(); const float* s10w = nx();
  BNp s11bn = rdbn(); const float* s11w = nx();
  BNp s2b1 = rdbn(), s2b2 = rdbn(); CVp s2c1 = rdcv(true), s2c2 = rdcv(true);
  BNp s3b1 = rdbn(), s3b2 = rdbn(); CVp s3c1 = rdcv(true), s3c2 = rdcv(true);
  BNp snorm1 = rdbn(), snorm2 = rdbn();
  CVp sres1 = rdcv(true), sres2 = rdcv(true);

  // ---------------- workspace ----------------
  char* wsb = (char*)d_ws; size_t wo = 0;
  auto alloc = [&](size_t bytes) -> float* {
    float* p = (float*)(wsb + wo);
    wo = (wo + bytes + 255) & ~(size_t)255;
    return p;
  };
  float* SC = alloc(2048 * 4);
  float* SH = alloc(2048 * 4);

  auto prep = [&](const BNp& bn, const float* cb, int C) {
    bn_prep<<<dim3((C + 255) / 256), dim3(256), 0, stream>>>(
        bn.g, bn.b, bn.m, bn.v, cb, SC, SH, C);
  };
  auto gemm = [&](const GP& p, int z, int mode) {
    const int cols = (mode == 2) ? 256 : 128;
    dim3 grid((unsigned)((p.N + cols - 1) / cols), (unsigned)(p.M / 16), (unsigned)z);
    if (mode == 0)      gemm_bn_act<0, 2><<<grid, dim3(128), 0, stream>>>(p);
    else if (mode == 1) gemm_bn_act<1, 2><<<grid, dim3(128), 0, stream>>>(p);
    else                gemm_bn_act<2, 4><<<grid, dim3(128), 0, stream>>>(p);
  };
  // conv/linear via implicit-GEMM; bn==null -> plain bias
  auto conv = [&](const float* in, float* out, const CVp& cv, const BNp* bn,
                  int Cin, int Cout, int IH, int IW, int KH, int strd, int pad,
                  int gelu, const float* res, _Float16* h16 = nullptr) {
    int OH = (IH + 2 * pad - KH) / strd + 1;
    int OW = (IW + 2 * pad - KH) / strd + 1;
    GP p{};
    p.A = cv.w; p.Bf = in;
    if (bn) { prep(*bn, cv.b, Cout); p.scale = SC; p.shift = SH; }
    else    { p.scale = nullptr;     p.shift = cv.b; }
    p.res = res; p.C = out; p.h16 = h16;
    p.M = Cout; p.N = 32 * OH * OW; p.K = Cin * KH * KH;
    p.aMod = 1;
    p.Cin = Cin; p.IH = IH; p.IW = IW;
    p.strd = strd; p.pad = pad; p.OW = OW; p.OHW = OH * OW;
    p.outMode = 0; p.gelu = gelu;
    gemm(p, 1, (KH == 3) ? 0 : 1);
  };
  // 1x1 GEMM over fragment-swizzled f16 B
  auto gemm_sw = [&](const _Float16* bsw, float* out, _Float16* out_sw,
                     const CVp& cv, const BNp* bn, int K, int M, int NG,
                     int Nspatial, int gelu, const float* res) {
    GP p{};
    p.A = cv.w; p.Bh = bsw;
    if (bn) { prep(*bn, cv.b, M); p.scale = SC; p.shift = SH; }
    else    { p.shift = cv.b; }
    p.res = res; p.C = out; p.h16 = out_sw;
    p.M = M; p.N = NG; p.K = K;
    p.aMod = 1; p.OHW = Nspatial; p.outMode = 0; p.gelu = gelu;
    gemm(p, 1, 2);
  };

  // activations
  float* T0 = alloc((size_t)32 * 32 * 128 * 128 * 4);
  float* T1 = alloc((size_t)32 * 32 * 128 * 128 * 4);
  float* T2 = alloc((size_t)32 * 64 * 64 * 64 * 4);
  float* T3 = alloc((size_t)32 * 64 * 64 * 64 * 4);
  float* T4 = alloc((size_t)32 * 64 * 64 * 64 * 4);
  float* T5 = alloc((size_t)32 * 128 * 32 * 32 * 4);
  float* T6 = alloc((size_t)32 * 128 * 32 * 32 * 4);
  float* T7 = alloc((size_t)32 * 128 * 32 * 32 * 4);
  float* P1 = alloc((size_t)32 * 128 * 16 * 16 * 4);
  float* X0 = alloc((size_t)32 * 256 * 16 * 16 * 4);
  float* FEAT = alloc((size_t)32 * 512 * 256 * 4);
  float* YB   = alloc((size_t)32 * 512 * 256 * 4);
  float* SXv = alloc(32 * 256 * 4); float* RXv = alloc(32 * 256 * 4);
  float* SYv = alloc(32 * 256 * 4); float* RYv = alloc(32 * 256 * 4);
  int*   IDX = (int*)alloc((size_t)32 * 256 * KNN * 4);
  _Float16* MSG = (_Float16*)alloc((size_t)32 * 512 * 256 * KNN * 2);
  float* H4   = alloc((size_t)32 * 512 * 256 * KNN * 4);
  float* HM   = alloc((size_t)32 * 512 * 256 * 4);
  float* GOUT = alloc((size_t)32 * 512 * 256 * 4);
  float* FF1  = alloc((size_t)32 * 2048 * 256 * 4);
  // fragment-swizzled f16 side buffers
  _Float16* SWH = (_Float16*)alloc((size_t)32 * 1024 * 256 * 2);   // 2C x NG
  _Float16* SWG = (_Float16*)alloc((size_t)32 * 512 * 256 * 2);    // C x NG
  _Float16* SWF = (_Float16*)alloc((size_t)32 * 2048 * 256 * 2);   // 4C x NG
  _Float16* X0sw   = (_Float16*)alloc((size_t)256 * 8192 * 2);
  _Float16* XT1sw  = (_Float16*)alloc((size_t)256 * 8192 * 2);
  _Float16* XT2asw = (_Float16*)alloc((size_t)384 * 2048 * 2);
  _Float16* XT2bsw = (_Float16*)alloc((size_t)384 * 2048 * 2);
  _Float16* XT3asw = (_Float16*)alloc((size_t)512 * 512 * 2);
  _Float16* XT3bsw = (_Float16*)alloc((size_t)512 * 512 * 2);
  float* XT1 = alloc(8u << 20);  float* X1 = alloc(8u << 20);
  float* XT2a = alloc(8u << 20); float* XT2b = alloc(8u << 20); float* X2 = alloc(8u << 20);
  float* XT3a = alloc(8u << 20); float* XT3b = alloc(8u << 20); float* X3 = alloc(8u << 20);
  float* XG  = alloc(32 * 512 * 4); float* XGb = alloc(32 * 512 * 4);
  float* XG2 = alloc(32 * 512 * 4);
  float* LX = alloc(32 * 512 * 4);
  float* GX = alloc(32 * 1536 * 4); float* GH = alloc(32 * 1536 * 4);
  float* HH = alloc(32 * 256 * 4);

  // ---------------- stem ----------------
  CVp c10{nullptr, s10w}, c11{nullptr, s11w};
  conv(input, T0, c10, &s10bn, 1, 32, 128, 128, 3, 1, 1, 1, nullptr);
  conv(T0, T1, c11, &s11bn, 32, 32, 128, 128, 3, 1, 1, 1, nullptr);
  conv(T1, T2, sres1, &snorm1, 32, 64, 128, 128, 1, 2, 0, 0, nullptr);   // tmp
  conv(T1, T3, s2c1, &s2b1, 32, 64, 128, 128, 3, 2, 1, 1, nullptr);
  conv(T3, T4, s2c2, &s2b2, 64, 64, 64, 64, 3, 1, 1, 1, T2);            // x
  conv(T4, T5, s3c1, &s3b1, 64, 128, 64, 64, 3, 2, 1, 1, nullptr);
  conv(T5, T6, s3c2, &s3b2, 128, 128, 32, 32, 3, 1, 1, 1, nullptr);     // out
  conv(T4, T7, sres2, &snorm2, 64, 128, 64, 64, 1, 2, 0, 0, T6);        // out+res2
  // ---------------- patchify ----------------
  conv(T7, P1, pf_c1, &pf_b1, 128, 128, 32, 32, 3, 2, 1, 1, nullptr);
  conv(P1, X0, pf_c2, &pf_b2, 128, 256, 16, 16, 3, 1, 1, 0, nullptr);
  add_bcast_sw<<<dim3((32 * 256 * 256 + 255) / 256), dim3(256), 0, stream>>>(
      X0, pos_embed, X0sw, (long long)32 * 256 * 256, 256 * 256, 256);
  mean_hw<<<dim3((32 * 256 + 255) / 256), dim3(256), 0, stream>>>(X0, XG, 32 * 256, 256);

  // ---------------- grapher+ffn block ----------------
  auto run_block = [&](const float* xin, const _Float16* xin_sw,
                       float* xout, _Float16* xout_sw, const BlkP& bp,
                       int C, int H, int W, int r) {
    int N = H * W, NG = 32 * N;
    int Mm = (r > 1) ? (H / r) * (W / r) : N;
    // fc1 + bn (B = swizzled block input)
    gemm_sw(xin_sw, FEAT, nullptr, bp.gf1, &bp.gf1bn, C, C, NG, N, 0, nullptr);
    const float* Y = FEAT;
    if (r > 1) {
      int tot = 32 * C * Mm;
      avgpool_k<<<dim3((tot + 255) / 256), dim3(256), 0, stream>>>(FEAT, YB, 32 * C, H, W, r);
      Y = YB;
    }
    col_norms<<<dim3((32 * N + 255) / 256), dim3(256), 0, stream>>>(FEAT, SXv, RXv, 32, C, N);
    col_norms<<<dim3((32 * Mm + 255) / 256), dim3(256), 0, stream>>>(Y, SYv, RYv, 32, C, Mm);
    knn_topk<<<dim3((32 * N + 255) / 256), dim3(256), 0, stream>>>(
        FEAT, Y, bp.relpos, SXv, RXv, SYv, RYv, 32, C, N, Mm, IDX);
    long long mt = (long long)32 * 2 * C * N * KNN;
    gather_msg<<<dim3((unsigned)((mt + 255) / 256)), dim3(256), 0, stream>>>(
        FEAT, Y, IDX, MSG, 32, C, N, Mm);
    // grouped EdgeConv as batched WMMA GEMM, z = (b,g), G=4
    int Cg = (2 * C) / 4, NK = N * KNN;
    prep(bp.gbn, bp.gconv.b, 2 * C);
    GP p{};
    p.A = bp.gconv.w; p.Bh = MSG; p.scale = SC; p.shift = SH; p.C = H4;
    p.M = Cg; p.N = NK; p.K = Cg;
    p.aMod = 4; p.aStride = (long long)Cg * Cg;
    p.bStrideZ = (long long)Cg * NK;
    p.outMode = 1; p.cStrideZ = (long long)Cg * NK;
    p.gelu = 1;
    gemm(p, 32 * 4, 2);
    long long ht = (long long)32 * 2 * C * N;
    maxk_k<<<dim3((unsigned)((ht + 255) / 256)), dim3(256), 0, stream>>>(
        H4, HM, SWH, 2 * C, N, ht);
    // fc2 + bn + shortcut; FFN
    gemm_sw(SWH, GOUT, SWG, bp.gf2, &bp.gf2bn, 2 * C, C, NG, N, 0, xin);
    gemm_sw(SWG, FF1, SWF, bp.f1, &bp.f1bn, C, 4 * C, NG, N, 1, nullptr);
    gemm_sw(SWF, xout, xout_sw, bp.f2, &bp.f2bn, 4 * C, C, NG, N, 0, GOUT);
  };

  // ---------------- backbone ----------------
  run_block(X0, X0sw, XT1, XT1sw, bk0, 256, 16, 16, 1);
  run_block(XT1, XT1sw, X1, X0sw /*scratch*/, bk1, 256, 16, 16, 1);
  conv(X1, XT2a, dn2.cv, &dn2.bn, 256, 384, 16, 16, 3, 2, 1, 0, nullptr, XT2asw);
  run_block(XT2a, XT2asw, XT2b, XT2bsw, bk3, 384, 8, 8, 2);
  run_block(XT2b, XT2bsw, X2, XT2asw /*scratch*/, bk4, 384, 8, 8, 2);
  conv(X2, XT3a, dn5.cv, &dn5.bn, 384, 512, 8, 8, 3, 2, 1, 0, nullptr, XT3asw);
  run_block(XT3a, XT3asw, XT3b, XT3bsw, bk6, 512, 4, 4, 1);
  run_block(XT3b, XT3bsw, X3, XT3asw /*scratch*/, bk7, 512, 4, 4, 1);

  // ---------------- GRU aggregation + head ----------------
  auto linear_ = [&](const float* in, float* out, const CVp& w,
                     int Cin, int Cout, int gelu) {
    conv(in, out, w, nullptr, Cin, Cout, 1, 1, 1, 1, 0, gelu, nullptr);
  };
  auto gru_stage = [&](const float* e, int L, int HW, const GRp& gr, const CVp& cn,
                       int Lout, const float* xg_in, float* xg_out) {
    mean_hw<<<dim3((32 * L + 255) / 256), dim3(256), 0, stream>>>(e, LX, 32 * L, HW);
    linear_(LX, GX, gr.x2h, L, 3 * L, 0);
    linear_(xg_in, GH, gr.h2h, L, 3 * L, 0);
    gru_add_k<<<dim3((32 * L + 255) / 256), dim3(256), 0, stream>>>(xg_in, GX, GH, XG2, 32, L);
    linear_(XG2, xg_out, cn, L, Lout, 1);
  };
  gru_stage(X1, 256, 256, gr0, con0, 384, XG, XGb);
  gru_stage(X2, 384, 64, gr1, con1, 512, XGb, XG);
  gru_stage(X3, 512, 16, gr2, con2, 512, XG, XGb);
  linear_(XGb, HH, pred0, 512, 256, 1);
  linear_(HH, (float*)d_out, pred1, 256, 16, 0);
}
